// CMN_18073222381847
// MI455X (gfx1250) — compile-verified, matrix operand order checked by reading, not run on hardware
//
#include <hip/hip_runtime.h>
#include <hip/hip_fp16.h>

// ---------------------------------------------------------------------------
// CDNA5 (gfx1250, wave32) implementation of the CMN reference.
//   T=2048 L=50 V=30000 D=300 NF=64 H=100 NC=7 WIND=40 HOPS=3
// Heavy GEMMs use v_wmma_f32_16x16x32_f16 with f32 accumulate.
// A-tiles are staged to LDS with global_load_async_to_lds_b128 (ASYNCcnt).
// ---------------------------------------------------------------------------

typedef __attribute__((ext_vector_type(16))) _Float16 v16h;
typedef __attribute__((ext_vector_type(8)))  float    v8f;

#define WMMA_F16(a, b, c) \
  __builtin_amdgcn_wmma_f32_16x16x32_f16(false, (a), false, (b), (short)0, (c), false, false)

// Flip to 0 if the async-to-LDS inline asm is rejected by the assembler.
#ifndef USE_ASYNC_LDS
#define USE_ASYNC_LDS 1
#endif

union FragG { v16h v; uint4 q[2]; };   // two aligned 16B loads
union FragL { v16h v; uint2 d[4]; };   // four aligned 8B LDS loads

__device__ __forceinline__ float sigf(float x) { return 1.0f / (1.0f + expf(-x)); }

// ---------------------------------------------------------------------------
// Pack f32 weights into WMMA B-panel layout (f16).
// Panel layout: for (kb, nt): 32 lanes x 16 halves, element (lane, i) holds
//   B[k][n], n = nt*16 + (lane&15), k = kb*32 + (i<8?0:16) + (lane>>4)*8 + (i&7)
// srcNK=1 -> src indexed [n*Kreal + k]  (i.e. source is (N,K), transpose)
// srcNK=0 -> src indexed [k*Nreal + n]  (source is (K,N))
// ---------------------------------------------------------------------------
__global__ void pack_b_kernel(const float* __restrict__ src, unsigned short* __restrict__ dst,
                              int Kreal, int Nreal, int Kp, int ntiles, int srcNK)
{
  long total = (long)(Kp >> 5) * ntiles * 32 * 16;
  for (long idx = blockIdx.x * (long)blockDim.x + threadIdx.x; idx < total;
       idx += (long)gridDim.x * blockDim.x) {
    int i    = (int)(idx & 15);
    int lane = (int)((idx >> 4) & 31);
    int blk  = (int)(idx >> 9);
    int nt   = blk % ntiles;
    int kb   = blk / ntiles;
    int n = nt * 16 + (lane & 15);
    int k = kb * 32 + ((i < 8) ? 0 : 16) + ((lane >> 4) << 3) + (i & 7);
    float v = 0.0f;
    if (k < Kreal && n < Nreal)
      v = srcNK ? src[(size_t)n * Kreal + k] : src[(size_t)k * Nreal + n];
    __half h = __float2half(v);
    dst[idx] = *reinterpret_cast<unsigned short*>(&h);
  }
}

__global__ void zero_u32_kernel(unsigned int* __restrict__ p, long n)
{
  for (long i = blockIdx.x * (long)blockDim.x + threadIdx.x; i < n;
       i += (long)gridDim.x * blockDim.x)
    p[i] = 0u;
}

// ---------------------------------------------------------------------------
// Generic WMMA GEMM: C[M x Nout] = act( A[M x KP](f16) * Bpanels + bias )
// A row r starts at A + r*rowStrideH + rowOffH (halves). KP is a compile-time
// constant (128 or 192) so the K loop fully unrolls, the prefetch predicate
// folds away, and two independent accumulators break WMMA->WMMA RAW chains.
// The block's 16-row A tile is staged once into LDS via
// global_load_async_to_lds_b128; all 4 waves (one n-tile each) then read
// their A fragments from LDS (ds_load_b128) -> 4x less global A traffic.
// ---------------------------------------------------------------------------
template <int KP>
__global__ __launch_bounds__(128) void gemm_f16_wmma(
    const __half* __restrict__ A, int rowStrideH, int rowOffH,
    const unsigned short* __restrict__ Bpan, const float* __restrict__ bias,
    float* __restrict__ C, __half* __restrict__ Ch, int chStrideH,
    int M, int Nout, int ntiles, int act)
{
  constexpr int KB = KP >> 5;          // 4 or 6 (always even)
  __shared__ __half aTile[16 * KP];
  int tid = threadIdx.x;
  int mtile = blockIdx.x;

  // --- cooperative A-tile staging (16B chunks; trip count divisible by 128)
  constexpr int CPR = KP >> 3;         // 16B chunks per row
  constexpr int CHUNKS = CPR << 4;     // 16 rows
#pragma unroll
  for (int c = tid; c < CHUNKS; c += 128) {
    int row = c / CPR;
    int col = (c - row * CPR) << 3;    // halves
    const __half* gsrc = A + (size_t)(mtile * 16 + row) * rowStrideH + rowOffH + col;
    __half* ldst = &aTile[row * KP + col];
#if USE_ASYNC_LDS
    // Low 32 bits of a flat pointer into LDS are the LDS byte offset
    // (aperture layout, ISA 10.2), which is what VDST expects.
    unsigned ldsOff = (unsigned)(uintptr_t)ldst;
    asm volatile("global_load_async_to_lds_b128 %0, %1, off"
                 :: "v"(ldsOff), "v"(gsrc) : "memory");
#else
    *(uint4*)ldst = *(const uint4*)gsrc;
#endif
  }
#if USE_ASYNC_LDS
  asm volatile("s_wait_asynccnt 0x0" ::: "memory");
#endif
  __syncthreads();

  int wave = tid >> 5, lane = tid & 31;
  int nt = blockIdx.y * 4 + wave;
  if (nt < ntiles) {
    int gg = lane >> 4, nl = lane & 15;
    v8f acc0 = {}, acc1 = {};
#pragma unroll
    for (int kb = 0; kb < KB; kb += 2) {
      FragG a0, b0, a1, b1;
      const __half* ap0 = &aTile[nl * KP + kb * 32 + gg * 8];
      a0.q[0] = *(const uint4*)(ap0);
      a0.q[1] = *(const uint4*)(ap0 + 16);
      a1.q[0] = *(const uint4*)(ap0 + 32);
      a1.q[1] = *(const uint4*)(ap0 + 48);
      const unsigned short* bp0 = Bpan + ((size_t)(kb * ntiles + nt) * 32 + lane) * 16;
      const unsigned short* bp1 = Bpan + ((size_t)((kb + 1) * ntiles + nt) * 32 + lane) * 16;
      b0.q[0] = *(const uint4*)(bp0);
      b0.q[1] = *(const uint4*)(bp0 + 8);
      b1.q[0] = *(const uint4*)(bp1);
      b1.q[1] = *(const uint4*)(bp1 + 8);
      if (kb + 2 < KB)  // folds at compile time (full unroll)
        __builtin_prefetch(Bpan + ((size_t)((kb + 2) * ntiles + nt) * 32 + lane) * 16, 0, 3);
      acc0 = WMMA_F16(a0.v, b0.v, acc0);   // independent accumulators:
      acc1 = WMMA_F16(a1.v, b1.v, acc1);   // no D->C RAW between neighbors
    }
    int n = nt * 16 + nl;
    if (n < Nout) {
      float bv = bias ? bias[n] : 0.0f;
#pragma unroll
      for (int j = 0; j < 8; ++j) {
        int m = mtile * 16 + gg * 8 + j;
        if (m < M) {
          float v = acc0[j] + acc1[j] + bv;
          if (act == 1) v = tanhf(v);
          C[(size_t)m * Nout + n] = v;
          if (Ch) Ch[(size_t)m * chStrideH + n] = __float2half(v);
        }
      }
    }
  }
}

// ---------------------------------------------------------------------------
// Conv feature extraction: one block per sentence. Sentence embeddings are
// gathered into LDS as f16; the A matrix of each conv-GEMM is a banded view
// (row p = halves [p*300, p*300+k*300)). Weights come pre-packed as B-panels.
// Output: relu+bias, max over valid positions -> featsH[t][192] (f16).
// ---------------------------------------------------------------------------
#define CONV_ROWS 53  // 50 real + padding so padded M-tiles read in-bounds

template <int KW>
__device__ __forceinline__ void conv_width(const __half* sent,
                                           const unsigned short* __restrict__ Bp,
                                           const float* __restrict__ bc,
                                           __half* __restrict__ featsH,
                                           int t, int wave, int lane, int foff)
{
  constexpr int Kreal = KW * 300;
  constexpr int Kp = (Kreal + 31) & ~31;
  constexpr int KB = Kp >> 5;
  constexpr int Pv = 50 - KW + 1;  // valid positions
  int g = lane >> 4, nl = lane & 15;
  v8f acc[3] = {};
  for (int kb = 0; kb < KB; ++kb) {
    FragG bf;
    const unsigned short* bp = Bp + ((size_t)(kb * 4 + wave) * 32 + lane) * 16;
    bf.q[0] = *(const uint4*)(bp);
    bf.q[1] = *(const uint4*)(bp + 8);
#pragma unroll
    for (int mt = 0; mt < 3; ++mt) {
      FragL af;
      const __half* ap = sent + ((mt * 16 + nl) * 300 + kb * 32 + g * 8);
      af.d[0] = *(const uint2*)(ap);
      af.d[1] = *(const uint2*)(ap + 4);
      af.d[2] = *(const uint2*)(ap + 16);
      af.d[3] = *(const uint2*)(ap + 20);
      acc[mt] = WMMA_F16(af.v, bf.v, acc[mt]);
    }
  }
  float bv = bc[wave * 16 + nl];
  float pm = 0.0f;  // relu >= 0, so 0 is the identity for masked max
#pragma unroll
  for (int mt = 0; mt < 3; ++mt)
#pragma unroll
    for (int j = 0; j < 8; ++j) {
      int p = mt * 16 + g * 8 + j;
      float v = fmaxf(acc[mt][j] + bv, 0.0f);
      if (p < Pv) pm = fmaxf(pm, v);
    }
  pm = fmaxf(pm, __shfl_xor(pm, 16));
  if (lane < 16)
    featsH[(size_t)t * 192 + foff + wave * 16 + lane] = __float2half(pm);
}

__global__ __launch_bounds__(128) void conv_feats_kernel(
    const int* __restrict__ sents, const float* __restrict__ emb,
    const unsigned short* __restrict__ Bp3, const unsigned short* __restrict__ Bp4,
    const unsigned short* __restrict__ Bp5,
    const float* __restrict__ bc3, const float* __restrict__ bc4,
    const float* __restrict__ bc5, __half* __restrict__ featsH)
{
  __shared__ __half sent[CONV_ROWS * 300];
  int t = blockIdx.x;
  for (int e = threadIdx.x; e < CONV_ROWS * 300; e += 128) {
    int tok = e / 300, d = e - tok * 300;
    float v = 0.0f;
    if (tok < 50) v = emb[(size_t)sents[t * 50 + tok] * 300 + d];
    sent[e] = __float2half(v);
  }
  __syncthreads();
  int wave = threadIdx.x >> 5, lane = threadIdx.x & 31;
  conv_width<3>(sent, Bp3, bc3, featsH, t, wave, lane, 0);
  conv_width<4>(sent, Bp4, bc4, featsH, t, wave, lane, 64);
  conv_width<5>(sent, Bp5, bc5, featsH, t, wave, lane, 128);
}

// ---------------------------------------------------------------------------
// Sequential context GRU. gi (input-side gates + bih) precomputed by WMMA
// GEMM; only h@Whh^T stays in the 2047-step dependency chain.
// ---------------------------------------------------------------------------
__global__ __launch_bounds__(384) void gru_scan_kernel(
    const float* __restrict__ gi_all, const float* __restrict__ Whh,
    const float* __restrict__ bhh, float* __restrict__ hiddens)
{
  __shared__ float h[100];
  __shared__ float gh[300];
  int tid = threadIdx.x;
  if (tid < 100) h[tid] = 0.0f;
  __syncthreads();
  for (int t = 0; t < 2047; ++t) {
    if (tid < 300) {
      const float* wr = Whh + (size_t)tid * 100;
      float s = bhh[tid];
#pragma unroll 4
      for (int k = 0; k < 100; ++k) s += h[k] * wr[k];
      gh[tid] = s;
    }
    __syncthreads();
    if (tid < 100) {
      const float* gi = gi_all + (size_t)t * 300;
      float r  = sigf(gi[tid] + gh[tid]);
      float z  = sigf(gi[tid + 100] + gh[tid + 100]);
      float nw = tanhf(gi[tid + 200] + r * gh[tid + 200]);
      float hn = (1.0f - z) * nw + z * h[tid];
      hiddens[(size_t)t * 100 + tid] = hn;
      h[tid] = hn;
    }
    __syncthreads();
  }
}

// mem[q][s][:] = valid ? hiddens[1+q-40+s] : 0 ; f16, K padded to 128.
__global__ void mem_init_kernel(const float* __restrict__ hiddens, __half* __restrict__ memH)
{
  long total = (long)2048 * 40 * 128;
  for (long idx = blockIdx.x * (long)blockDim.x + threadIdx.x; idx < total;
       idx += (long)gridDim.x * blockDim.x) {
    int d = (int)(idx & 127);
    long row = idx >> 7;
    int s = (int)(row % 40);
    int q = (int)(row / 40);
    int src = 1 + q - 40 + s;
    float v = 0.0f;
    if (q < 2047 && d < 100 && src >= 0) v = hiddens[(size_t)src * 100 + d];
    memH[idx] = __float2half(v);
  }
}

__global__ void query_init_kernel(const float* __restrict__ s_utt, float* __restrict__ query)
{
  long total = (long)2047 * 100;
  for (long idx = blockIdx.x * (long)blockDim.x + threadIdx.x; idx < total;
       idx += (long)gridDim.x * blockDim.x) {
    int q = (int)(idx / 100), d = (int)(idx % 100);
    query[idx] = s_utt[(size_t)(1 + q) * 100 + d];
  }
}

// One attention hop: scores -> softmax -> query = tanh(query + a@mem).
__global__ __launch_bounds__(256) void attend_kernel(const __half* __restrict__ memH,
                                                     float* __restrict__ query)
{
  __shared__ float sc[40];
  __shared__ float aw[40];
  __shared__ float qs[100];
  __shared__ float ssum;
  int q = blockIdx.x;  // 0..2046
  int wave = threadIdx.x >> 5, lane = threadIdx.x & 31;
  if (threadIdx.x < 100) qs[threadIdx.x] = query[(size_t)q * 100 + threadIdx.x];
  __syncthreads();
  const __half* mq = memH + (size_t)q * 40 * 128;
  for (int s = wave; s < 40; s += 8) {
    float p = 0.0f;
    for (int d = lane; d < 100; d += 32) p += __half2float(mq[s * 128 + d]) * qs[d];
#pragma unroll
    for (int o = 16; o >= 1; o >>= 1) p += __shfl_xor(p, o);
    bool valid = (1 + q - 40 + s) >= 0;
    if (lane == 0) sc[s] = valid ? p : -1e10f;
  }
  __syncthreads();
  if (threadIdx.x == 0) {
    float mx = sc[0];
    for (int s = 1; s < 40; ++s) mx = fmaxf(mx, sc[s]);
    float sum = 0.0f;
    for (int s = 0; s < 40; ++s) { float e = expf(sc[s] - mx); aw[s] = e; sum += e; }
    ssum = sum;
  }
  __syncthreads();
  if (threadIdx.x < 100) {
    int d = threadIdx.x;
    float c = 0.0f;
    for (int s = 0; s < 40; ++s) c += aw[s] * __half2float(mq[s * 128 + d]);
    query[(size_t)q * 100 + d] = tanhf(qs[d] + c / ssum);
  }
}

// GRU gate math for one memory slot s across all 2047 queries.
__global__ void mem_gate_kernel(const float* __restrict__ gi, const float* __restrict__ gh,
                                float* __restrict__ hmS, __half* __restrict__ hmH,
                                __half* __restrict__ memH, int s)
{
  long total = (long)2047 * 100;
  for (long idx = blockIdx.x * (long)blockDim.x + threadIdx.x; idx < total;
       idx += (long)gridDim.x * blockDim.x) {
    int q = (int)(idx / 100), n = (int)(idx % 100);
    bool valid = (1 + q - 40 + s) >= 0;
    size_t b = (size_t)q * 300;
    float r  = sigf(gi[b + n] + gh[b + n]);
    float z  = sigf(gi[b + 100 + n] + gh[b + 100 + n]);
    float nw = tanhf(gi[b + 200 + n] + r * gh[b + 200 + n]);
    float hOld = hmS[(size_t)q * 100 + n];
    float hNew = (1.0f - z) * nw + z * hOld;
    float out = valid ? hNew : hOld;
    hmS[(size_t)q * 100 + n] = out;
    __half oh = __float2half(out);
    hmH[(size_t)q * 128 + n] = oh;
    memH[((size_t)q * 40 + s) * 128 + n] = oh;
  }
}

// logits = s_cont @ Wcls + bcls ; log_softmax over 7 classes.
__global__ __launch_bounds__(32) void classify_kernel(const float* __restrict__ s_utt,
                                                      const float* __restrict__ query,
                                                      const float* __restrict__ Wcls,
                                                      const float* __restrict__ bcls,
                                                      float* __restrict__ out)
{
  __shared__ float lg[7];
  int t = blockIdx.x;
  const float* row = (t == 0) ? s_utt : (query + (size_t)(t - 1) * 100);
  int n = threadIdx.x;
  if (n < 7) {
    float s = bcls[n];
    for (int k = 0; k < 100; ++k) s += row[k] * Wcls[k * 7 + n];
    lg[n] = s;
  }
  __syncthreads();
  if (n < 7) {
    float mx = lg[0];
    for (int j = 1; j < 7; ++j) mx = fmaxf(mx, lg[j]);
    float sum = 0.0f;
    for (int j = 0; j < 7; ++j) sum += expf(lg[j] - mx);
    out[(size_t)t * 7 + n] = lg[n] - mx - logf(sum);
  }
}

// ---------------------------------------------------------------------------
extern "C" void kernel_launch(void* const* d_in, const int* in_sizes, int n_in,
                              void* d_out, int out_size, void* d_ws, size_t ws_size,
                              hipStream_t stream)
{
  (void)in_sizes; (void)n_in; (void)out_size; (void)ws_size;
  const int*   sents = (const int*)d_in[0];
  // d_in[1] = lengths (unused by reference)
  const float* emb   = (const float*)d_in[2];
  const float* Wc3   = (const float*)d_in[3];
  const float* bc3   = (const float*)d_in[4];
  const float* Wc4   = (const float*)d_in[5];
  const float* bc4   = (const float*)d_in[6];
  const float* Wc5   = (const float*)d_in[7];
  const float* bc5   = (const float*)d_in[8];
  const float* Wt    = (const float*)d_in[9];
  const float* bt    = (const float*)d_in[10];
  const float* Wih_c = (const float*)d_in[11];
  const float* Whh_c = (const float*)d_in[12];
  const float* bih_c = (const float*)d_in[13];
  const float* bhh_c = (const float*)d_in[14];
  const float* Wih_m = (const float*)d_in[15];
  const float* Whh_m = (const float*)d_in[16];
  const float* bih_m = (const float*)d_in[17];
  const float* bhh_m = (const float*)d_in[18];
  const float* Wcls  = (const float*)d_in[19];
  const float* bcls  = (const float*)d_in[20];

  char* ws = (char*)d_ws;
  size_t off = 0;
  auto take = [&](size_t bytes) -> char* {
    char* p = ws + off;
    off = (off + bytes + 255) & ~(size_t)255;
    return p;
  };
  // Packed f16 B-panels (1024 B per (kb, ntile) block)
  unsigned short* Bp3   = (unsigned short*)take((size_t)29 * 4 * 1024);  // Kp=928
  unsigned short* Bp4   = (unsigned short*)take((size_t)38 * 4 * 1024);  // Kp=1216
  unsigned short* Bp5   = (unsigned short*)take((size_t)47 * 4 * 1024);  // Kp=1504
  unsigned short* BpWt  = (unsigned short*)take((size_t)6 * 7 * 1024);   // Kp=192
  unsigned short* BpIhc = (unsigned short*)take((size_t)4 * 19 * 1024);  // Kp=128
  unsigned short* BpIhm = (unsigned short*)take((size_t)4 * 19 * 1024);
  unsigned short* BpHhm = (unsigned short*)take((size_t)4 * 19 * 1024);
  __half* featsH = (__half*)take((size_t)2048 * 192 * 2);
  float*  s_utt  = (float*)take((size_t)2048 * 100 * 4);
  __half* s_uttH = (__half*)take((size_t)2048 * 128 * 2);
  float*  gi_all = (float*)take((size_t)2048 * 300 * 4);
  float*  hidden = (float*)take((size_t)2048 * 100 * 4);
  __half* memH   = (__half*)take((size_t)81920 * 128 * 2);  // (2048*40) x 128
  float*  query  = (float*)take((size_t)2048 * 100 * 4);
  float*  hmS    = (float*)take((size_t)2048 * 100 * 4);
  __half* hmH    = (__half*)take((size_t)2048 * 128 * 2);
  float*  gi_m   = (float*)take((size_t)2048 * 300 * 4);
  float*  gh_m   = (float*)take((size_t)2048 * 300 * 4);

  auto packB = [&](const float* src, unsigned short* dst, int Kr, int Nr, int Kp, int nt,
                   int srcNK) {
    long total = (long)(Kp >> 5) * nt * 512;
    int grid = (int)((total + 255) / 256);
    pack_b_kernel<<<grid, 256, 0, stream>>>(src, dst, Kr, Nr, Kp, nt, srcNK);
  };
  // Conv weights are (NF, k*D) row-major -> (N,K) source, transpose into B.
  packB(Wc3, Bp3, 900, 64, 928, 4, 1);
  packB(Wc4, Bp4, 1200, 64, 1216, 4, 1);
  packB(Wc5, Bp5, 1500, 64, 1504, 4, 1);
  packB(Wt, BpWt, 192, 100, 192, 7, 0);        // Wt is (K,N)
  packB(Wih_c, BpIhc, 100, 300, 128, 19, 1);   // gi = x @ W^T
  packB(Wih_m, BpIhm, 100, 300, 128, 19, 1);
  packB(Whh_m, BpHhm, 100, 300, 128, 19, 1);

  auto zero = [&](void* p, long u32s) {
    int grid = (int)((u32s + 255) / 256);
    if (grid > 16384) grid = 16384;
    zero_u32_kernel<<<grid, 256, 0, stream>>>((unsigned int*)p, u32s);
  };
  auto gemm = [&](const __half* A, int rs, int ro, const unsigned short* Bp, const float* bias,
                  float* C, __half* Ch, int chs, int M, int Kp, int Nout, int nt, int act) {
    dim3 g((M + 15) / 16, (nt + 3) / 4);
    if (Kp == 192)
      gemm_f16_wmma<192><<<g, 128, 0, stream>>>(A, rs, ro, Bp, bias, C, Ch, chs, M, Nout, nt, act);
    else
      gemm_f16_wmma<128><<<g, 128, 0, stream>>>(A, rs, ro, Bp, bias, C, Ch, chs, M, Nout, nt, act);
  };

  // 1) conv features (gather + 3 banded WMMA GEMMs + relu/max-pool)
  conv_feats_kernel<<<2048, 128, 0, stream>>>(sents, emb, Bp3, Bp4, Bp5, bc3, bc4, bc5, featsH);

  // 2) s_utt = tanh(feats @ Wt + bt); f16 copy padded to K=128
  zero(s_uttH, (long)2048 * 128 / 2);
  gemm(featsH, 192, 0, BpWt, bt, s_utt, s_uttH, 128, 2048, 192, 100, 7, 1);

  // 3) gi_all = s_utt @ Wih_c^T + bih_c (hoisted input gates)
  gemm(s_uttH, 128, 0, BpIhc, bih_c, gi_all, nullptr, 0, 2048, 128, 300, 19, 0);

  // 4) sequential context GRU
  gru_scan_kernel<<<1, 384, 0, stream>>>(gi_all, Whh_c, bhh_c, hidden);

  // 5) memory windows + query init
  mem_init_kernel<<<16384, 256, 0, stream>>>(hidden, memH);
  query_init_kernel<<<800, 256, 0, stream>>>(s_utt, query);

  // 6) 3 attention hops; memory-GRU update after hops 0 and 1 only
  //    (the reference's final-hop update is dead code — its result is unused).
  for (int hop = 0; hop < 3; ++hop) {
    attend_kernel<<<2047, 256, 0, stream>>>(memH, query);
    if (hop < 2) {
      zero(hmS, (long)2048 * 100);
      zero(hmH, (long)2048 * 128 / 2);
      for (int s = 0; s < 40; ++s) {
        gemm(memH, 40 * 128, s * 128, BpIhm, bih_m, gi_m, nullptr, 0, 2047, 128, 300, 19, 0);
        gemm(hmH, 128, 0, BpHhm, bhh_m, gh_m, nullptr, 0, 2047, 128, 300, 19, 0);
        mem_gate_kernel<<<800, 256, 0, stream>>>(gi_m, gh_m, hmS, hmH, memH, s);
      }
    }
  }

  // 7) classifier + log_softmax
  classify_kernel<<<2048, 32, 0, stream>>>(s_utt, query, Wcls, bcls, (float*)d_out);
}
